// Inception_49632642072786
// MI455X (gfx1250) — 16-bit and 8-bit matrix operand order repaired and checked by reading, compiled, not run on hardware
//
#include <hip/hip_runtime.h>
#include <hip/hip_bf16.h>
#include <cstdint>

typedef __attribute__((ext_vector_type(16))) _Float16 v16h;
typedef __attribute__((ext_vector_type(8)))  _Float16 v8h;
typedef __attribute__((ext_vector_type(8)))  float    v8f;
typedef __attribute__((ext_vector_type(8)))  int      v8i;
typedef __attribute__((ext_vector_type(4)))  int      i32x4;

constexpr int kB    = 32;
constexpr int kCin  = 480;
constexpr int kH    = 28;
constexpr int kW    = 28;
constexpr int kHW   = 784;
constexpr int kM    = kB * kHW;
constexpr int kCout = 512;

#if defined(__has_builtin)
#  if __has_builtin(__builtin_amdgcn_global_load_async_to_lds_b128) && \
      __has_builtin(__builtin_amdgcn_s_wait_asynccnt)
#    define USE_ASYNC_LDS 1
#  endif
#endif
#ifndef USE_ASYNC_LDS
#  define USE_ASYNC_LDS 0
#endif

__device__ __forceinline__ void copy16_g2lds(const void* g, void* lds) {
#if USE_ASYNC_LDS
  __builtin_amdgcn_global_load_async_to_lds_b128(
      (__attribute__((address_space(1))) i32x4*)g,
      (__attribute__((address_space(3))) i32x4*)lds, 0, 0);
#else
  *(uint4*)lds = *(const uint4*)g;
#endif
}
__device__ __forceinline__ void g2lds_fence() {
#if USE_ASYNC_LDS
  __builtin_amdgcn_s_wait_asynccnt(0);
#endif
}

__global__ void pack_x_kernel(const float* __restrict__ x,
                              _Float16* __restrict__ xT,
                              _Float16* __restrict__ pT) {
  int idx = blockIdx.x * 256 + threadIdx.x;
  if (idx >= kB * kCin * kHW) return;
  int hw = idx % kHW;
  int t  = idx / kHW;
  int c  = t % kCin;
  int n  = t / kCin;
  int h = hw / kW, w = hw % kW;
  const float* base = x + (size_t)(n * kCin + c) * kHW;
  float v = base[hw];
  float mx = -3.4e38f;
#pragma unroll
  for (int dy = -1; dy <= 1; ++dy) {
    int hh = h + dy;
    if (hh < 0 || hh >= kH) continue;
#pragma unroll
    for (int dx = -1; dx <= 1; ++dx) {
      int ww = w + dx;
      if (ww < 0 || ww >= kW) continue;
      mx = fmaxf(mx, base[hh * kW + ww]);
    }
  }
  size_t m = (size_t)n * kHW + hw;
  xT[m * kCin + c] = (_Float16)v;
  pT[m * kCin + c] = (_Float16)mx;
}

__global__ void quantw_f16_kernel(const float* __restrict__ w,
                                  _Float16* __restrict__ o, int n) {
  int i = blockIdx.x * 256 + threadIdx.x;
  if (i >= n) return;
  float q = rintf(w[i] * 128.0f);
  q = fminf(fmaxf(q, -127.0f), 127.0f);
  o[i] = (_Float16)(q * 0.0078125f);
}

__global__ void quantw2b_kernel(const float* __restrict__ w,
                                signed char* __restrict__ o) {
  int idx = blockIdx.x * 256 + threadIdx.x;
  if (idx >= 9 * 256 * 128) return;
  int ci  = idx & 127;
  int tmp = idx >> 7;
  int co  = tmp & 255;
  int tap = tmp >> 8;
  signed char q = 0;
  if (co < 208 && ci < 96) {
    float f = rintf(w[(size_t)(co * 96 + ci) * 9 + tap] * 128.0f);
    f = fminf(fmaxf(f, -127.0f), 127.0f);
    q = (signed char)(int)f;
  }
  o[idx] = q;
}

__global__ void quantw3b_kernel(const float* __restrict__ w,
                                signed char* __restrict__ o) {
  int idx = blockIdx.x * 256 + threadIdx.x;
  if (idx >= 64 * 448) return;
  int kk = idx % 448;
  int co = idx / 448;
  int tap = kk >> 4, ci = kk & 15;
  signed char q = 0;
  if (co < 48 && tap < 25) {
    float f = rintf(w[(size_t)(co * 16 + ci) * 25 + tap] * 128.0f);
    f = fminf(fmaxf(f, -127.0f), 127.0f);
    q = (signed char)(int)f;
  }
  o[idx] = q;
}

__global__ void __launch_bounds__(256)
gemm_f16_q(const _Float16* __restrict__ A, const _Float16* __restrict__ Bw,
           int Ncols, int mode, int cOff, int ldo,
           float* __restrict__ out, unsigned char* __restrict__ actOut) {
  __shared__ _Float16 smA[128 * 40];
  __shared__ _Float16 smB[64 * 40];
  const int m0 = blockIdx.x * 128;
  const int n0 = blockIdx.y * 64;
  const int t = threadIdx.x;
  const int lane = t & 31, wave = t >> 5;
  const int lm = lane & 15, hi = lane >> 4;

  v8f acc0 = {}, acc1 = {}, acc2 = {}, acc3 = {};

  const int rowA = t >> 1, halfA = t & 1;
  const int rowB = t >> 2, qB = t & 3;
  const _Float16* gA = A + (size_t)(m0 + rowA) * kCin + halfA * 16;
  const int cB = n0 + rowB;
  const _Float16* gB = Bw + (size_t)cB * kCin + qB * 8;

  for (int k0 = 0; k0 < kCin; k0 += 32) {
    __syncthreads();
    {
      _Float16* d = smA + rowA * 40 + halfA * 16;
      copy16_g2lds(gA + k0, d);
      copy16_g2lds(gA + k0 + 8, d + 8);
      if (k0 + 32 < kCin) __builtin_prefetch(gA + k0 + 32, 0, 0);
    }
    {
      _Float16* d = smB + rowB * 40 + qB * 8;
      if (cB < Ncols) copy16_g2lds(gB + k0, d);
      else            *(uint4*)d = make_uint4(0u, 0u, 0u, 0u);
    }
    g2lds_fence();
    __syncthreads();

    v8h alo = *(const v8h*)(smA + (wave * 16 + lm) * 40 + hi * 8);
    v8h ahi = *(const v8h*)(smA + (wave * 16 + lm) * 40 + 16 + hi * 8);
    v16h a;
#pragma unroll
    for (int i = 0; i < 8; ++i) { a[i] = alo[i]; a[i + 8] = ahi[i]; }

#pragma unroll
    for (int ns = 0; ns < 4; ++ns) {
      v8h blo = *(const v8h*)(smB + (ns * 16 + lm) * 40 + hi * 8);
      v8h bhi = *(const v8h*)(smB + (ns * 16 + lm) * 40 + 16 + hi * 8);
      v16h b;
#pragma unroll
      for (int i = 0; i < 8; ++i) { b[i] = blo[i]; b[i + 8] = bhi[i]; }
      v8f* accp = (ns == 0) ? &acc0 : (ns == 1) ? &acc1 : (ns == 2) ? &acc2 : &acc3;
      *accp = __builtin_amdgcn_wmma_f32_16x16x32_f16(false, a, false, b,
                                                     (short)0, *accp, false, false);
    }
  }

#pragma unroll
  for (int ns = 0; ns < 4; ++ns) {
    const v8f& acc = (ns == 0) ? acc0 : (ns == 1) ? acc1 : (ns == 2) ? acc2 : acc3;
    int c = n0 + ns * 16 + lm;
#pragma unroll
    for (int r = 0; r < 8; ++r) {
      int m = m0 + wave * 16 + r + hi * 8;
      float q = rintf(acc[r] * 128.0f);
      q = fminf(fmaxf(q, 0.0f), 127.0f);
      if (mode == 0) {
        if (c < Ncols) {
          int n = m / kHW, hw = m % kHW;
          __builtin_nontemporal_store(
              q * 0.0078125f,
              out + (size_t)n * (kCout * kHW) + (size_t)(cOff + c) * kHW + hw);
        }
      } else {
        if (c < ldo) actOut[(size_t)m * ldo + c] = (unsigned char)(int)q;
      }
    }
  }
}

__global__ void __launch_bounds__(256)
gemm3x3_iu8(const unsigned char* __restrict__ act, const signed char* __restrict__ Bw,
            float* __restrict__ out) {
  __shared__ unsigned char smA[128 * 80];
  __shared__ unsigned char smB[64 * 80];
  const int m0 = blockIdx.x * 128;
  const int n0 = blockIdx.y * 64;
  const int t = threadIdx.x;
  const int lane = t & 31, wave = t >> 5;
  const int lm = lane & 15, hi = lane >> 4;

  v8i acc0 = {}, acc1 = {}, acc2 = {}, acc3 = {};

  const int rowA = t >> 1, halfA = t & 1;
  const int rowB = t >> 2, qB = t & 3;

  const int mA = m0 + rowA;
  const int nImg = mA / kHW;
  const int hwA = mA - nImg * kHW;
  const int hA = hwA / kW;
  const int wA = hwA - hA * kW;

  for (int tap = 0; tap < 9; ++tap) {
    const int dy = tap / 3 - 1, dx = tap % 3 - 1;
    const int hh = hA + dy, ww = wA + dx;
    const bool ok = hh >= 0 && hh < kH && ww >= 0 && ww < kW;
    const int srcPix = nImg * kHW + hh * kW + ww;
    for (int kb = 0; kb < 2; ++kb) {
      __syncthreads();
      {
        unsigned char* d = smA + rowA * 80 + halfA * 32;
        if (ok) {
          const unsigned char* s = act + (size_t)srcPix * 128 + kb * 64 + halfA * 32;
          copy16_g2lds(s, d);
          copy16_g2lds(s + 16, d + 16);
        } else {
          uint4 z = make_uint4(0u, 0u, 0u, 0u);
          *(uint4*)d = z;
          *(uint4*)(d + 16) = z;
        }
      }
      {
        copy16_g2lds(Bw + ((size_t)tap * 256 + (n0 + rowB)) * 128 + kb * 64 + qB * 16,
                     smB + rowB * 80 + qB * 16);
      }
      g2lds_fence();
      __syncthreads();

      const unsigned char* ar = smA + (wave * 16 + lm) * 80 + hi * 16;
      uint4 a0 = *(const uint4*)(ar);
      uint4 a1 = *(const uint4*)(ar + 32);
      v8i a = {(int)a0.x, (int)a0.y, (int)a0.z, (int)a0.w,
               (int)a1.x, (int)a1.y, (int)a1.z, (int)a1.w};

#pragma unroll
      for (int ns = 0; ns < 4; ++ns) {
        const unsigned char* br = smB + (ns * 16 + lm) * 80 + hi * 16;
        uint4 b0 = *(const uint4*)(br);
        uint4 b1 = *(const uint4*)(br + 32);
        v8i b = {(int)b0.x, (int)b0.y, (int)b0.z, (int)b0.w,
                 (int)b1.x, (int)b1.y, (int)b1.z, (int)b1.w};
        v8i* accp = (ns == 0) ? &acc0 : (ns == 1) ? &acc1 : (ns == 2) ? &acc2 : &acc3;
        *accp = __builtin_amdgcn_wmma_i32_16x16x64_iu8(false, a, true, b, *accp, false, false);
      }
    }
  }

#pragma unroll
  for (int ns = 0; ns < 4; ++ns) {
    const v8i& acc = (ns == 0) ? acc0 : (ns == 1) ? acc1 : (ns == 2) ? acc2 : acc3;
    int c = n0 + ns * 16 + lm;
    if (c < 208) {
#pragma unroll
      for (int r = 0; r < 8; ++r) {
        int m = m0 + wave * 16 + r + hi * 8;
        float q = rintf((float)acc[r] * 0.0078125f);
        q = fminf(fmaxf(q, 0.0f), 127.0f);
        int n = m / kHW, hw = m % kHW;
        __builtin_nontemporal_store(
            q * 0.0078125f,
            out + (size_t)n * (kCout * kHW) + (size_t)(192 + c) * kHW + hw);
      }
    }
  }
}

__global__ void __launch_bounds__(256)
gemm5x5_iu8(const unsigned char* __restrict__ act, const signed char* __restrict__ Bw,
            float* __restrict__ out) {
  __shared__ unsigned char smA[128 * 80];
  __shared__ unsigned char smB[64 * 80];
  const int m0 = blockIdx.x * 128;
  const int t = threadIdx.x;
  const int lane = t & 31, wave = t >> 5;
  const int lm = lane & 15, hi = lane >> 4;

  v8i acc0 = {}, acc1 = {}, acc2 = {};
  const int rowA = t >> 1, halfA = t & 1;
  const int rowB = t >> 2, qB = t & 3;

  const int mA = m0 + rowA;
  const int nImg = mA / kHW;
  const int hwA = mA - nImg * kHW;
  const int hA = hwA / kW;
  const int wA = hwA - hA * kW;

  for (int ks = 0; ks < 448; ks += 64) {
    __syncthreads();
    {
#pragma unroll
      for (int s = 0; s < 2; ++s) {
        int kk = ks + halfA * 32 + s * 16;
        int tap = kk >> 4;
        unsigned char* d = smA + rowA * 80 + halfA * 32 + s * 16;
        bool ld = false;
        int srcPix = 0;
        if (tap < 25) {
          int dy = tap / 5 - 2, dx = tap % 5 - 2;
          int hh = hA + dy, ww = wA + dx;
          if (hh >= 0 && hh < kH && ww >= 0 && ww < kW) {
            ld = true;
            srcPix = nImg * kHW + hh * kW + ww;
          }
        }
        if (ld) copy16_g2lds(act + (size_t)srcPix * 16, d);
        else    *(uint4*)d = make_uint4(0u, 0u, 0u, 0u);
      }
    }
    {
      copy16_g2lds(Bw + (size_t)rowB * 448 + ks + qB * 16,
                   smB + rowB * 80 + qB * 16);
    }
    g2lds_fence();
    __syncthreads();

    const unsigned char* ar = smA + (wave * 16 + lm) * 80 + hi * 16;
    uint4 a0 = *(const uint4*)(ar);
    uint4 a1 = *(const uint4*)(ar + 32);
    v8i a = {(int)a0.x, (int)a0.y, (int)a0.z, (int)a0.w,
             (int)a1.x, (int)a1.y, (int)a1.z, (int)a1.w};

#pragma unroll
    for (int ns = 0; ns < 3; ++ns) {
      const unsigned char* br = smB + (ns * 16 + lm) * 80 + hi * 16;
      uint4 b0 = *(const uint4*)(br);
      uint4 b1 = *(const uint4*)(br + 32);
      v8i b = {(int)b0.x, (int)b0.y, (int)b0.z, (int)b0.w,
               (int)b1.x, (int)b1.y, (int)b1.z, (int)b1.w};
      v8i* accp = (ns == 0) ? &acc0 : (ns == 1) ? &acc1 : &acc2;
      *accp = __builtin_amdgcn_wmma_i32_16x16x64_iu8(false, a, true, b, *accp, false, false);
    }
  }

#pragma unroll
  for (int ns = 0; ns < 3; ++ns) {
    const v8i& acc = (ns == 0) ? acc0 : (ns == 1) ? acc1 : acc2;
    int c = ns * 16 + lm;
#pragma unroll
    for (int r = 0; r < 8; ++r) {
      int m = m0 + wave * 16 + r + hi * 8;
      float q = rintf((float)acc[r] * 0.0078125f);
      q = fminf(fmaxf(q, 0.0f), 127.0f);
      int n = m / kHW, hw = m % kHW;
      __builtin_nontemporal_store(
          q * 0.0078125f,
          out + (size_t)n * (kCout * kHW) + (size_t)(400 + c) * kHW + hw);
    }
  }
}

extern "C" void kernel_launch(void* const* d_in, const int* in_sizes, int n_in,
                              void* d_out, int out_size, void* d_ws, size_t ws_size,
                              hipStream_t stream) {
  const float* x   = (const float*)d_in[0];
  const float* w1  = (const float*)d_in[1];
  const float* w2a = (const float*)d_in[2];
  const float* w2b = (const float*)d_in[3];
  const float* w3a = (const float*)d_in[4];
  const float* w3b = (const float*)d_in[5];
  const float* w4  = (const float*)d_in[6];
  float* out = (float*)d_out;
  (void)in_sizes; (void)n_in; (void)out_size; (void)ws_size;

  auto align256 = [](size_t v) { return (v + 255) & ~(size_t)255; };
  char* p = (char*)d_ws;
  _Float16* xT = (_Float16*)p;              p += align256((size_t)kM * kCin * 2);
  _Float16* pT = (_Float16*)p;              p += align256((size_t)kM * kCin * 2);
  unsigned char* act2a = (unsigned char*)p; p += align256((size_t)kM * 128);
  unsigned char* act3a = (unsigned char*)p; p += align256((size_t)kM * 16);
  _Float16* wq1  = (_Float16*)p;            p += align256((size_t)192 * kCin * 2);
  _Float16* wq2a = (_Float16*)p;            p += align256((size_t)96 * kCin * 2);
  _Float16* wq3a = (_Float16*)p;            p += align256((size_t)16 * kCin * 2);
  _Float16* wq4  = (_Float16*)p;            p += align256((size_t)64 * kCin * 2);
  signed char* wq2b = (signed char*)p;      p += align256((size_t)9 * 256 * 128);
  signed char* wq3b = (signed char*)p;      p += align256((size_t)64 * 448);

  {
    int tot = kB * kCin * kHW;
    pack_x_kernel<<<(tot + 255) / 256, 256, 0, stream>>>(x, xT, pT);
  }
  quantw_f16_kernel<<<(192 * kCin + 255) / 256, 256, 0, stream>>>(w1, wq1, 192 * kCin);
  quantw_f16_kernel<<<(96 * kCin + 255) / 256, 256, 0, stream>>>(w2a, wq2a, 96 * kCin);
  quantw_f16_kernel<<<(16 * kCin + 255) / 256, 256, 0, stream>>>(w3a, wq3a, 16 * kCin);
  quantw_f16_kernel<<<(64 * kCin + 255) / 256, 256, 0, stream>>>(w4, wq4, 64 * kCin);
  quantw2b_kernel<<<(9 * 256 * 128 + 255) / 256, 256, 0, stream>>>(w2b, wq2b);
  quantw3b_kernel<<<(64 * 448 + 255) / 256, 256, 0, stream>>>(w3b, wq3b);

  dim3 blk(256);
  const int mg = kM / 128;
  gemm_f16_q<<<dim3(mg, 2), blk, 0, stream>>>(xT, wq2a, 96, 1, 0, 128, out, act2a);
  gemm_f16_q<<<dim3(mg, 1), blk, 0, stream>>>(xT, wq3a, 16, 1, 0, 16, out, act3a);
  gemm_f16_q<<<dim3(mg, 3), blk, 0, stream>>>(xT, wq1, 192, 0, 0, 0, out, nullptr);
  gemm_f16_q<<<dim3(mg, 1), blk, 0, stream>>>(pT, wq4, 64, 0, 448, 0, out, nullptr);
  gemm3x3_iu8<<<dim3(mg, 4), blk, 0, stream>>>(act2a, wq2b, out);
  gemm5x5_iu8<<<dim3(mg, 1), blk, 0, stream>>>(act3a, wq3b, out);
}
